// PoincareProbe_23691039605211
// MI455X (gfx1250) — matrix operand order repaired and checked by reading, not compiled, not run on hardware
//
#include <hip/hip_runtime.h>
#include <cstdint>

typedef _Float16 f16;
typedef f16   v16h __attribute__((ext_vector_type(16)));
typedef float v8f  __attribute__((ext_vector_type(8)));
typedef float v4f  __attribute__((ext_vector_type(4)));

#define H_DIM 768
#define D_DIM 64
#define S_LEN 1024
#define B_SZ  64
#define ROWS  128                    // rows (of B*S) per workgroup
#define KSTEP 32
#define KB_COUNT (H_DIM / KSTEP)     // 24
#define NT_COUNT (D_DIM / 16)        // 4
#define XS_STRIDE 80                 // LDS f16 tile row stride in bytes (64B data + 16B pad)
#define BPACK_ELEMS (KB_COUNT * NT_COUNT * 32 * 8)   // 24576 dwords

__device__ __forceinline__ float artanh_c(float x) {
    x = fminf(x, 1.0f - 1e-5f);
    x = fmaxf(x, -1.0f + 1e-5f);
    return 0.5f * (log1pf(x) - log1pf(-x));
}

// ||mobius_add(x,y)|| given x2=||x||^2, xy=<x,y>, y2=||y||^2 -> 2*artanh(||.||)
__device__ __forceinline__ float hyp_dist(float x2, float xy, float y2) {
    float a   = 1.0f + 2.0f * xy + y2;
    float c   = 1.0f - x2;
    float den = fmaxf(1.0f + 2.0f * xy + x2 * y2, 1e-15f);
    float nvsq = fmaxf(a * a * x2 + 2.0f * a * c * xy + c * c * y2, 0.0f);
    float nv   = fmaxf(sqrtf(nvsq) / den, 1e-15f);
    return 2.0f * artanh_c(nv);
}

// Repack proj [768x64] (row-major f32) into WMMA B-fragment layout (f16):
// bp[kb][nt][lane][v] : lanes 0-15 -> N=lane,    V_v = K{2v,2v+1} (v<4), K{8+2v,9+2v} (v>=4);
//                       lanes 16-31 -> N=lane-16, V_v = K{8+2v,9+2v} (v<4), K{16+2v,17+2v} (v>=4)  (K base = kb*32)
__global__ void pack_proj_kernel(const float* __restrict__ proj, uint32_t* __restrict__ bp) {
    int idx = blockIdx.x * blockDim.x + threadIdx.x;
    if (idx >= BPACK_ELEMS) return;
    int v    = idx & 7;
    int lane = (idx >> 3) & 31;
    int nt   = (idx >> 8) & 3;
    int kb   = idx >> 10;
    int n  = nt * 16 + (lane & 15);
    int k0 = kb * KSTEP + ((v & 4) ? 16 : 0) + ((lane & 16) ? 8 : 0) + 2 * (v & 3);
    union { f16 h[2]; uint32_t u; } cv;
    cv.h[0] = (f16)proj[(size_t)k0 * D_DIM + n];
    cv.h[1] = (f16)proj[(size_t)(k0 + 1) * D_DIM + n];
    bp[idx] = cv.u;
}

__global__ void __launch_bounds__(256) poincare_main_kernel(
    const float* __restrict__ X, const uint32_t* __restrict__ bp,
    const float* __restrict__ trans, const float* __restrict__ pos,
    const float* __restrict__ neg, float* __restrict__ partials)
{
    __shared__ __align__(16) unsigned char xs[ROWS * XS_STRIDE];  // f16 A tile (padded)
    __shared__ float ts[ROWS * 65];                               // f32 t tile (padded)
    __shared__ float trans_s[D_DIM * D_DIM];
    __shared__ float ps[D_DIM];
    __shared__ float ns[D_DIM];
    __shared__ float wred[16];

    const int tid  = threadIdx.x;
    const int lane = tid & 31;
    const int w    = tid >> 5;                      // wave 0..7 -> rows w*16..w*16+15
    const size_t m0 = (size_t)blockIdx.x * ROWS;

    for (int i = tid; i < D_DIM * D_DIM; i += 256) trans_s[i] = trans[i];
    if (tid < D_DIM) { ps[tid] = pos[tid]; ns[tid] = neg[tid]; }

    v8f acc[NT_COUNT];
    {
        v8f z = {0,0,0,0,0,0,0,0};
        #pragma unroll
        for (int nt = 0; nt < NT_COUNT; ++nt) acc[nt] = z;
    }

    const int srow = tid >> 3;                      // staging: 32 rows/pass, 8 lanes/row
    const int sc4  = tid & 7;
    const int aRow = w * 16 + (lane & 15);
    const unsigned char* abase = xs + aRow * XS_STRIDE + ((lane & 16) ? 16 : 0);

    for (int kb = 0; kb < KB_COUNT; ++kb) {
        __syncthreads();
        // ---- stage 128x32 fp32 -> fp16 into LDS; single-pass stream => non-temporal ----
        #pragma unroll
        for (int pass = 0; pass < 4; ++pass) {
            int row = pass * 32 + srow;
            const v4f xv = __builtin_nontemporal_load(
                (const v4f*)(X + (m0 + row) * H_DIM + kb * KSTEP + sc4 * 4));
            union { f16 h[4]; uint2 u; } cv;
            cv.h[0] = (f16)xv.x; cv.h[1] = (f16)xv.y;
            cv.h[2] = (f16)xv.z; cv.h[3] = (f16)xv.w;
            *(uint2*)(xs + row * XS_STRIDE + sc4 * 8) = cv.u;
        }
        __syncthreads();
        // ---- A fragment per ISA 16-bit 16x32 layout ----
        union { uint4 q[2]; v16h v; } af;
        af.q[0] = *(const uint4*)(abase);        // lanes<16: K0..7  ; lanes>=16: K8..15
        af.q[1] = *(const uint4*)(abase + 32);   // lanes<16: K16..23; lanes>=16: K24..31
        #pragma unroll
        for (int nt = 0; nt < NT_COUNT; ++nt) {
            union { uint4 q[2]; v16h v; } bfr;
            const uint32_t* bq = bp + (((size_t)kb * NT_COUNT + nt) * 32 + lane) * 8;
            bfr.q[0] = *(const uint4*)(bq);
            bfr.q[1] = *(const uint4*)(bq + 4);
            acc[nt] = __builtin_amdgcn_wmma_f32_16x16x32_f16(
                false, af.v, false, bfr.v, (short)0, acc[nt], false, false);
        }
    }

    // ---- spill C tiles to LDS: VGPR r, lanes<16 -> (M=r,N=lane); lanes>=16 -> (M=r+8,N=lane-16)
    {
        int mofs = w * 16 + ((lane & 16) ? 8 : 0);
        int nl = lane & 15;
        #pragma unroll
        for (int nt = 0; nt < NT_COUNT; ++nt)
            #pragma unroll
            for (int r = 0; r < 8; ++r)
                ts[(mofs + r) * 65 + nt * 16 + nl] = acc[nt][r];
    }
    __syncthreads();

    // ---- per-row hyperbolic math in fp32, one thread per row (waves 0-3 fully active) ----
    float dp = 0.0f, dn = 0.0f;
    if (tid < ROWS) {
        float u[D_DIM];
        #pragma unroll
        for (int k = 0; k < D_DIM; ++k) u[k] = ts[tid * 65 + k];
        float nsq = 0.0f;
        #pragma unroll
        for (int k = 0; k < D_DIM; ++k) nsq = fmaf(u[k], u[k], nsq);
        float nrm = fmaxf(sqrtf(nsq), 1e-15f);
        float es  = tanhf(nrm) / nrm;                 // expmap0 scale
        #pragma unroll
        for (int k = 0; k < D_DIM; ++k) u[k] *= es;   // u = expmap0(t)
        float xn = fmaxf(tanhf(nrm), 1e-15f);         // ||expmap0(t)||

        // mx = u @ trans^T ; only need ||mx||^2, <pos,mx>, <neg,mx>
        float mxsq = 0.0f, dpm = 0.0f, dnm = 0.0f;
        for (int j = 0; j < D_DIM; ++j) {
            const float* tr = &trans_s[j * D_DIM];
            float m = 0.0f;
            #pragma unroll
            for (int k = 0; k < D_DIM; ++k) m = fmaf(u[k], tr[k], m);
            mxsq = fmaf(m, m, mxsq);
            dpm  = fmaf(ps[j], m, dpm);
            dnm  = fmaf(ns[j], m, dnm);
        }
        float x2p = 0.0f, x2n = 0.0f;
        #pragma unroll
        for (int j = 0; j < D_DIM; ++j) {
            x2p = fmaf(ps[j], ps[j], x2p);
            x2n = fmaf(ns[j], ns[j], x2n);
        }
        float mxn = fmaxf(sqrtf(mxsq), 1e-15f);
        float scale = (mxsq == 0.0f) ? 0.0f
                      : tanhf(mxn / xn * artanh_c(xn)) / mxn;   // res = scale*mx (0 if mx==0)
        float y2 = scale * scale * mxsq;
        // dist(p, res): x = -p  =>  <x,res> = -scale*<p,mx>
        dp = hyp_dist(x2p, -scale * dpm, y2);   // dist(pos,.) -> neg_logits -> out col 0
        dn = hyp_dist(x2n, -scale * dnm, y2);   // dist(neg,.) -> pos_logits -> out col 1
    }

    // ---- deterministic block reduction ----
    #pragma unroll
    for (int off = 16; off > 0; off >>= 1) {
        dp += __shfl_xor(dp, off, 32);
        dn += __shfl_xor(dn, off, 32);
    }
    if (lane == 0) { wred[w * 2] = dp; wred[w * 2 + 1] = dn; }
    __syncthreads();
    if (tid == 0) {
        float sp = 0.0f, sn = 0.0f;
        #pragma unroll
        for (int i = 0; i < 8; ++i) { sp += wred[i * 2]; sn += wred[i * 2 + 1]; }
        partials[blockIdx.x * 2 + 0] = sp;
        partials[blockIdx.x * 2 + 1] = sn;
    }
}

__global__ void finalize_kernel(const float* __restrict__ partials, float* __restrict__ out) {
    int idx = threadIdx.x;              // 0..127 : b = idx>>1, c = idx&1
    if (idx >= B_SZ * 2) return;
    int b = idx >> 1, c = idx & 1;
    float s = 0.0f;
    #pragma unroll
    for (int i = 0; i < 8; ++i)         // 8 row-blocks per batch, fixed order
        s += partials[((b * 8 + i) << 1) + c];
    out[idx] = s;
}

extern "C" void kernel_launch(void* const* d_in, const int* in_sizes, int n_in,
                              void* d_out, int out_size, void* d_ws, size_t ws_size,
                              hipStream_t stream) {
    (void)in_sizes; (void)n_in; (void)out_size; (void)ws_size;
    const float* X     = (const float*)d_in[0];   // [64,1024,768]
    const float* proj  = (const float*)d_in[1];   // [768,64]
    const float* trans = (const float*)d_in[2];   // [64,64]
    const float* pos   = (const float*)d_in[3];   // [64]
    const float* neg   = (const float*)d_in[4];   // [64]
    float* out = (float*)d_out;                   // [64,2]

    uint32_t* bp = (uint32_t*)d_ws;                                    // 96 KB B-pack
    float* partials = (float*)((char*)d_ws + (size_t)BPACK_ELEMS * 4); // 512*2 floats

    pack_proj_kernel<<<(BPACK_ELEMS + 255) / 256, 256, 0, stream>>>(proj, bp);
    poincare_main_kernel<<<(B_SZ * S_LEN) / ROWS, 256, 0, stream>>>(
        X, bp, trans, pos, neg, partials);
    finalize_kernel<<<1, 128, 0, stream>>>(partials, out);
}
